// TransformerMatchModule_60052232733127
// MI455X (gfx1250) — matrix operand order repaired and checked by reading, not compiled, not run on hardware
//
#include <hip/hip_runtime.h>
#include <math.h>

// ---------------- CDNA5 WMMA types ----------------
typedef __attribute__((ext_vector_type(16))) __bf16 v16bf;
typedef __attribute__((ext_vector_type(8)))  __bf16 v8bf;
typedef __attribute__((ext_vector_type(8)))  float  v8f;

__device__ __forceinline__ __bf16 f2bf(float f) {
  unsigned int u = __float_as_uint(f);
  u = (u + 0x7FFFu + ((u >> 16) & 1u)) >> 16;   // round-to-nearest-even
  unsigned short s = (unsigned short)u;
  __bf16 r; __builtin_memcpy(&r, &s, 2);
  return r;
}

union BFV { uint4 u[2]; v16bf v; };

// unguarded 16-halfword fragment from two 16B chunks
__device__ __forceinline__ v16bf ld16(const __bf16* p0, const __bf16* p1) {
  BFV t;
  t.u[0] = *(const uint4*)p0;
  t.u[1] = *(const uint4*)p1;
  return t.v;
}
// first chunk loaded, second chunk zero (A-fragment, 16-remainder case)
__device__ __forceinline__ v16bf ld16_z1(const __bf16* p0) {
  BFV t;
  t.u[0] = *(const uint4*)p0;
  t.u[1] = make_uint4(0u,0u,0u,0u);
  return t.v;
}
// both chunks loaded then zeroed per-lane (B-fragment, 16-remainder: hi lanes invalid)
__device__ __forceinline__ v16bf ld16_sel(const __bf16* p0, const __bf16* p1, bool ok) {
  BFV t;
  const uint4 z = make_uint4(0u,0u,0u,0u);
  uint4 a = *(const uint4*)p0;
  uint4 b = *(const uint4*)p1;
  t.u[0] = ok ? a : z;
  t.u[1] = ok ? b : z;
  return t.v;
}

// ---------------- generic batched bf16 WMMA GEMM ----------------
// C[z][M,N] = A[z][M,K] * W[z][N,K]^T (+ bias[n]),  z = zb*H + zh with separate strides.
// One wave computes a 16x32 output slab (two 16x16 WMMA tiles sharing the A fragment).
struct GemmDesc {
  const __bf16* A; const __bf16* W; float* C; const float* bias;
  int M, N, K, lda, ldw, ldc, H;
  long long aB, aH, wB, wH, cB, cH;   // element strides
};

__global__ void __launch_bounds__(256) gemm_bf16_wmma(GemmDesc d) {
  const int wave = threadIdx.x >> 5;
  const int lane = threadIdx.x & 31;
  const int l16  = lane & 15;
  const bool hi  = lane >= 16;
  const int ntN = d.N >> 4;
  const int np  = blockIdx.y * 8 + wave;       // n-tile-pair index
  const int nt0 = np * 2;
  if (nt0 >= ntN) return;
  const bool has1 = (nt0 + 1) < ntN;
  const int mt = blockIdx.x;
  const int z  = blockIdx.z;
  const int zb = z / d.H, zh = z - zb * d.H;
  const __bf16* A = d.A + zb * d.aB + zh * d.aH;
  const __bf16* W = d.W + zb * d.wB + zh * d.wH;
  float* C = d.C + zb * d.cB + zh * d.cH;
  const int m0 = mt << 4;
  const int n0 = nt0 << 4;
  // A 16x32 bf16 layout: lane<16 -> K {k0..k0+7, k0+16..k0+23}; lane>=16 -> {k0+8.., k0+24..}
  const __bf16* arow = A + (long long)(m0 + l16) * d.lda + (hi ? 8 : 0);
  // B 32x16 layout: lane<16 -> K {k0..k0+7, k0+16..k0+23}; lane>=16 -> {k0+8.., k0+24..}
  const __bf16* w0 = W + (long long)(n0 + l16) * d.ldw + (hi ? 8 : 0);
  const __bf16* w1 = W + (long long)((has1 ? n0 + 16 : n0) + l16) * d.ldw + (hi ? 8 : 0);
  v8f acc0 = {0.f,0.f,0.f,0.f,0.f,0.f,0.f,0.f};
  v8f acc1 = {0.f,0.f,0.f,0.f,0.f,0.f,0.f,0.f};
  const int kFull = d.K & ~31;
  for (int k0 = 0; k0 < kFull; k0 += 32) {
    v16bf a  = ld16(arow + k0, arow + k0 + 16);
    v16bf b0 = ld16(w0 + k0, w0 + k0 + 16);
    v16bf b1 = ld16(w1 + k0, w1 + k0 + 16);
    acc0 = __builtin_amdgcn_wmma_f32_16x16x32_bf16(false, a, false, b0, (short)0, acc0, false, false);
    acc1 = __builtin_amdgcn_wmma_f32_16x16x32_bf16(false, a, false, b1, (short)0, acc1, false, false);
  }
  if (d.K & 16) {   // uniform remainder of exactly 16 (all K here are multiples of 16)
    v16bf a  = ld16_z1(arow + kFull);                       // second 8-chunk always K-OOB
    v16bf b0 = ld16_z1(w0 + kFull);                         // second 8-chunk always K-OOB
    v16bf b1 = ld16_z1(w1 + kFull);
    acc0 = __builtin_amdgcn_wmma_f32_16x16x32_bf16(false, a, false, b0, (short)0, acc0, false, false);
    acc1 = __builtin_amdgcn_wmma_f32_16x16x32_bf16(false, a, false, b1, (short)0, acc1, false, false);
  }
  if (d.bias) {
    const float bv0 = d.bias[n0 + l16];
    const float bv1 = d.bias[(has1 ? n0 + 16 : n0) + l16];
    #pragma unroll
    for (int r = 0; r < 8; ++r) { acc0[r] += bv0; acc1[r] += bv1; }
  }
  // C/D layout: VGPR r: lanes 0-15 -> M=m0+r, lanes 16-31 -> M=m0+8+r; N=n0+(lane&15)
  float* crow = C + (long long)(m0 + (hi ? 8 : 0)) * d.ldc + n0 + l16;
  #pragma unroll
  for (int r = 0; r < 8; ++r) crow[(long long)r * d.ldc] = acc0[r];
  if (has1) {
    #pragma unroll
    for (int r = 0; r < 8; ++r) crow[(long long)r * d.ldc + 16] = acc1[r];
  }
}

// ---------------- elementwise / reduction kernels ----------------
__global__ void cvt_kernel(const float* in, __bf16* out, long long n) {
  long long i = (long long)blockIdx.x * blockDim.x + threadIdx.x;
  const long long st = (long long)gridDim.x * blockDim.x;
  for (; i < n; i += st) out[i] = f2bf(in[i]);
}

// dw[b,i,j] = (1/(dist+0.01)) / rowsum
__global__ void __launch_bounds__(256) dist_kernel(const float* ctr, float* dw, int Kp) {
  const int b = blockIdx.y, i = blockIdx.x, j = threadIdx.x;
  const float* cb = ctr + (long long)b * Kp * 3;
  const float xi = cb[i*3+0], yi = cb[i*3+1], zi = cb[i*3+2];
  const float dx = cb[j*3+0]-xi, dy = cb[j*3+1]-yi, dz = cb[j*3+2]-zi;
  const float w = 1.f / (sqrtf(dx*dx + dy*dy + dz*dz) + 0.01f);
  __shared__ float red[256];
  red[j] = w; __syncthreads();
  for (int s = 128; s > 0; s >>= 1) { if (j < s) red[j] += red[j+s]; __syncthreads(); }
  dw[((long long)b * Kp + i) * Kp + j] = w / red[0];
}

__global__ void bn_prelu_kernel(float* X, __bf16* Xbf, const float* g, const float* b,
                                const float* rm, const float* rv, const float* alpha,
                                int perC, long long total, int C) {
  long long i = (long long)blockIdx.x * blockDim.x + threadIdx.x;
  if (i >= total) return;
  const int c = (int)(i % C);
  const float x = X[i];
  float y = (x - rm[c]) * rsqrtf(rv[c] + 1e-5f) * g[c] + b[c];
  const float a = perC ? alpha[c] : alpha[0];
  y = y > 0.f ? y : a * y;
  X[i] = y;
  if (Xbf) Xbf[i] = f2bf(y);
}

// out = LayerNorm( (resid? resid+proj : (relu? max(proj,0):proj)) )  over C=128
__global__ void __launch_bounds__(128) ln_kernel(const float* resid, const float* proj,
                                                 const float* g, const float* b,
                                                 float* out, __bf16* outbf, int doRelu) {
  const int t = blockIdx.x, c = threadIdx.x;
  const long long i = (long long)t * 128 + c;
  float v = proj[i];
  if (resid) v += resid[i];
  if (doRelu) v = fmaxf(v, 0.f);
  __shared__ float red[128];
  red[c] = v; __syncthreads();
  for (int s = 64; s > 0; s >>= 1) { if (c < s) red[c] += red[c+s]; __syncthreads(); }
  const float mu = red[0] * (1.f/128.f); __syncthreads();
  const float dv = v - mu;
  red[c] = dv * dv; __syncthreads();
  for (int s = 64; s > 0; s >>= 1) { if (c < s) red[c] += red[c+s]; __syncthreads(); }
  const float var = red[0] * (1.f/128.f);
  const float y = dv * rsqrtf(var + 1e-5f) * g[c] + b[c];
  out[i] = y;
  if (outbf) outbf[i] = f2bf(y);
}

// scale + optional dw-bias + optional mask + softmax, writing bf16 IN PLACE (row pitch stays nk floats)
__global__ void __launch_bounds__(256) softmax_kernel(float* S, int nk, int nq, float scale,
                                                      const float* dwAdd, int dwDiv, long long dwBStride,
                                                      const unsigned char* mask, int maskDiv) {
  const long long row = blockIdx.x;
  const int tid = threadIdx.x;
  float* srow = S + row * nk;
  const bool have = tid < nk;
  float x = 0.f;
  if (have) {
    x = srow[tid] * scale;
    if (dwAdd) x += dwAdd[(row / dwDiv) * dwBStride + (long long)(row % nq) * nk + tid];
    if (mask && !mask[(row / maskDiv) * nk + tid]) x = -__builtin_inff();
  }
  __shared__ float red[256];
  red[tid] = have ? x : -__builtin_inff(); __syncthreads();
  for (int s = 128; s > 0; s >>= 1) { if (tid < s) red[tid] = fmaxf(red[tid], red[tid+s]); __syncthreads(); }
  const float mx = red[0]; __syncthreads();
  const float e = have ? __expf(x - mx) : 0.f;
  red[tid] = e; __syncthreads();
  for (int s = 128; s > 0; s >>= 1) { if (tid < s) red[tid] += red[tid+s]; __syncthreads(); }
  const float inv = 1.f / red[0];
  __syncthreads();                         // all f32 reads done before bf16 overwrite
  if (have) ((__bf16*)srow)[tid] = f2bf(e * inv);
}

// VT[batch][h][d][k] = (bf16) V[(batch*nk + k)*ldv + h*dv + d]
__global__ void packT_kernel(const float* V, __bf16* VT, long long total,
                             int Hh, int dv, int nk, int ldv) {
  long long i = (long long)blockIdx.x * blockDim.x + threadIdx.x;
  if (i >= total) return;
  const int k = (int)(i % nk); long long r = i / nk;
  const int d = (int)(r % dv); r /= dv;
  const int h = (int)(r % Hh);
  const long long bb = r / Hh;
  VT[i] = f2bf(V[(bb * nk + k) * (long long)ldv + h * dv + d]);
}

// v0[bc][k][c] = x2[bc>>5][k][c]   (broadcast over CHUNK=32)
__global__ void bcast_kernel(const float* x2, float* v0, __bf16* v0bf, long long total) {
  long long i = (long long)blockIdx.x * blockDim.x + threadIdx.x;
  if (i >= total) return;
  const int c = (int)(i % 128);
  const long long t = i / 128;
  const int k = (int)(t % 256);
  const int bc = (int)(t / 256);
  const float v = x2[((long long)(bc >> 5) * 256 + k) * 128 + c];
  v0[i] = v; v0bf[i] = f2bf(v);
}

// out[t] = bias + sum_c X[t,c]*w[c]  (final 128 -> 1 head; one wave per token)
__global__ void __launch_bounds__(256) dot_kernel(const float* X, const float* w,
                                                  const float* bias, float* out, int T) {
  const int t = blockIdx.x * 8 + (threadIdx.x >> 5);
  const int lane = threadIdx.x & 31;
  if (t >= T) return;
  float s = 0.f;
  for (int c = lane; c < 128; c += 32) s += X[(long long)t * 128 + c] * w[c];
  for (int off = 16; off > 0; off >>= 1) s += __shfl_down(s, off, 32);
  if (lane == 0) out[t] = s + bias[0];
}

// ---------------- host orchestration ----------------
extern "C" void kernel_launch(void* const* d_in, const int* in_sizes, int n_in,
                              void* d_out, int out_size, void* d_ws, size_t ws_size,
                              hipStream_t stream) {
  (void)in_sizes; (void)out_size; (void)ws_size;
  if (n_in < 85) return;
  // jax pytree leaves: dicts flatten in sorted-key order
  enum {
    CA0_KB=0, CA0_KW, CA0_LNB, CA0_LNG, CA0_OB, CA0_OW, CA0_QB, CA0_QW, CA0_VB, CA0_VW,
    CA1_KB, CA1_KW, CA1_LNB, CA1_LNG, CA1_OB, CA1_OW, CA1_QB, CA1_QW, CA1_VB, CA1_VW,
    FC_A1, FC_BN1B, FC_BN1G, FC_BN1RM, FC_BN1RV, FC_C1B, FC_C1W, FC_C2B, FC_C2W,
    LF_LINB, LF_LINW, LF_LNB, LF_LNG,
    LSA_KB, LSA_KW, LSA_LNB, LSA_LNG, LSA_OB, LSA_OW, LSA_QB, LSA_QW, LSA_VB, LSA_VW,
    M_A1, M_A2, M_BN1B, M_BN1G, M_BN1RM, M_BN1RV, M_BN2B, M_BN2G, M_BN2RM, M_BN2RV,
    M_C1B, M_C1W, M_C2B, M_C2W, M_C3B, M_C3W,
    SA0_KB, SA0_KW, SA0_LNB, SA0_LNG, SA0_OB, SA0_OW, SA0_QB, SA0_QW, SA0_VB, SA0_VW,
    SA1_KB, SA1_KW, SA1_LNB, SA1_LNG, SA1_OB, SA1_OW, SA1_QB, SA1_QW, SA1_VB, SA1_VW,
    IN_FEATS, IN_CENTER, IN_LANG, IN_LMASK, IN_PBMASK, IN_ISTRAIN
  };
  auto F = [&](int i) { return (const float*)d_in[i]; };
  const unsigned char* lmask = (const unsigned char*)d_in[IN_LMASK];

  const long long T1 = 2048, TL = 20480, TV = 65536;
  char* wsb = (char*)d_ws; size_t cur = 0;
  auto alloc = [&](size_t bytes) -> void* {
    cur = (cur + 255) & ~(size_t)255;
    void* p = wsb + cur; cur += bytes + 256; return p;  // 256B tail pad (clamped over-reads)
  };

  float* VA  = (float*)alloc(TV*128*4);  float* VB  = (float*)alloc(TV*128*4);
  float* VQ  = (float*)alloc(TV*128*4);  float* VK  = (float*)alloc(TV*128*4);
  float* VV  = (float*)alloc(TV*128*4);  float* VT1 = (float*)alloc(TV*128*4);
  __bf16* bVA  = (__bf16*)alloc(TV*128*2); __bf16* bVB  = (__bf16*)alloc(TV*128*2);
  __bf16* bVQ  = (__bf16*)alloc(TV*128*2); __bf16* bVK  = (__bf16*)alloc(TV*128*2);
  __bf16* bVT1 = (__bf16*)alloc(TV*128*2);
  float*  Lf  = (float*)alloc(TL*128*4);  __bf16* Lbf = (__bf16*)alloc(TL*128*2);
  __bf16* VT  = (__bf16*)alloc((size_t)256*4*32*256*2);
  float*  DW  = (float*)alloc((size_t)8*256*256*4);
  float*  SC  = (float*)alloc((size_t)1024*256*256*4);

  auto cvta = [&](const float* in, __bf16* out, long long n) {
    int blocks = (int)((n + 255) / 256); if (blocks > 16384) blocks = 16384;
    cvt_kernel<<<blocks, 256, 0, stream>>>(in, out, n);
  };
  auto cvtw = [&](int idx, long long n) {
    __bf16* p = (__bf16*)alloc(n * 2);
    cvta(F(idx), p, n);
    return p;
  };
  auto gemm = [&](const __bf16* A, int lda, long long aB, long long aH,
                  const __bf16* W, int ldw, long long wB, long long wH,
                  float* C, int ldc, long long cB, long long cH,
                  const float* bias, int M, int N, int K, int Z, int Hh) {
    GemmDesc d; d.A=A; d.W=W; d.C=C; d.bias=bias; d.M=M; d.N=N; d.K=K;
    d.lda=lda; d.ldw=ldw; d.ldc=ldc; d.H=Hh;
    d.aB=aB; d.aH=aH; d.wB=wB; d.wH=wH; d.cB=cB; d.cH=cH;
    const int npairs = (N/16 + 1) / 2;
    dim3 grid(M/16, (npairs + 7) / 8, Z);
    gemm_bf16_wmma<<<grid, 256, 0, stream>>>(d);
  };
  auto lin = [&](const __bf16* A, int lda, const __bf16* W, const float* bias,
                 float* C, int M, int N, int K) {
    gemm(A, lda, 0, 0, W, K, 0, 0, C, N, 0, 0, bias, M, N, K, 1, 1);
  };
  auto ln = [&](const float* resid, const float* proj, int gi, int bi,
                float* out, __bf16* outbf, long long tokens, int relu) {
    ln_kernel<<<(unsigned)tokens, 128, 0, stream>>>(resid, proj, F(gi), F(bi), out, outbf, relu);
  };

  // ---- convert all weight matrices to bf16 ----
  __bf16 *wFC1=cvtw(FC_C1W,128*128), *wFC2=cvtw(FC_C2W,128*128);
  __bf16 *wS0Q=cvtw(SA0_QW,128*128), *wS0K=cvtw(SA0_KW,128*128), *wS0V=cvtw(SA0_VW,128*128), *wS0O=cvtw(SA0_OW,128*128);
  __bf16 *wLG =cvtw(LF_LINW,128*256);
  __bf16 *wLQ =cvtw(LSA_QW,64*128),  *wLK =cvtw(LSA_KW,64*128),  *wLV =cvtw(LSA_VW,64*128),  *wLO =cvtw(LSA_OW,128*64);
  __bf16 *wC0Q=cvtw(CA0_QW,128*128), *wC0K=cvtw(CA0_KW,128*128), *wC0V=cvtw(CA0_VW,128*128), *wC0O=cvtw(CA0_OW,128*128);
  __bf16 *wS1Q=cvtw(SA1_QW,128*128), *wS1K=cvtw(SA1_KW,128*128), *wS1V=cvtw(SA1_VW,128*128), *wS1O=cvtw(SA1_OW,128*128);
  __bf16 *wC1Q=cvtw(CA1_QW,128*128), *wC1K=cvtw(CA1_KW,128*128), *wC1V=cvtw(CA1_VW,128*128), *wC1O=cvtw(CA1_OW,128*128);
  __bf16 *wM1 =cvtw(M_C1W,128*128),  *wM2 =cvtw(M_C2W,128*128);

  const float SCL32 = 0.17677669529663687f;  // 1/sqrt(32)

  // ================= phase X: fc + sa0 on (8,256,128) =================
  cvta(F(IN_FEATS), bVA, T1*128);
  lin(bVA, 128, wFC1, F(FC_C1B), VT1, (int)T1, 128, 128);
  bn_prelu_kernel<<<(unsigned)((T1*128+255)/256),256,0,stream>>>(VT1, bVT1, F(FC_BN1G), F(FC_BN1B),
      F(FC_BN1RM), F(FC_BN1RV), F(FC_A1), 1, T1*128, 128);
  lin(bVT1, 128, wFC2, F(FC_C2B), VA, (int)T1, 128, 128);   // x
  cvta(VA, bVA, T1*128);
  lin(bVA, 128, wS0Q, F(SA0_QB), VQ, (int)T1, 128, 128);  cvta(VQ, bVQ, T1*128);
  lin(bVA, 128, wS0K, F(SA0_KB), VK, (int)T1, 128, 128);  cvta(VK, bVK, T1*128);
  lin(bVA, 128, wS0V, F(SA0_VB), VV, (int)T1, 128, 128);
  packT_kernel<<<(unsigned)((8LL*4*32*256+255)/256),256,0,stream>>>(VV, VT, 8LL*4*32*256, 4, 32, 256, 128);
  dist_kernel<<<dim3(256,8),256,0,stream>>>(F(IN_CENTER), DW, 256);
  // scores: z over 8*4, M=256,N=256,K=32
  gemm(bVQ,128, 256*128,32, bVK,128, 256*128,32, SC,256, 4LL*65536,65536, nullptr, 256,256,32, 32,4);
  softmax_kernel<<<32*256,256,0,stream>>>(SC, 256, 256, SCL32, DW, 1024, 65536, nullptr, 1);
  // att@v: A=(bf16)SC lda=512, W=VT ldw=256, C per (b,h) column block
  gemm((const __bf16*)SC,512, 4LL*256*512,256*512, VT,256, 4LL*8192,8192,
       VT1,128, 256*128,32, nullptr, 256,32,256, 32,4);
  cvta(VT1, bVT1, T1*128);
  lin(bVT1, 128, wS0O, F(SA0_OB), VQ, (int)T1, 128, 128);
  ln(VA, VQ, SA0_LNG, SA0_LNB, VB, bVB, T1, 0);             // x2 = sa0 out
  bcast_kernel<<<(unsigned)((TV*128+255)/256),256,0,stream>>>(VB, VA, bVA, TV*128);  // v0

  // ================= phase L: langfc + lsa on (256,80,128) =================
  cvta(F(IN_LANG), bVQ, TL*256);
  lin(bVQ, 256, wLG, F(LF_LINB), VT1, (int)TL, 128, 256);
  ln(nullptr, VT1, LF_LNG, LF_LNB, Lf, Lbf, TL, 1);         // relu + LN
  lin(Lbf, 128, wLQ, F(LSA_QB), VQ, (int)TL, 64, 128);  cvta(VQ, bVQ, TL*64);
  lin(Lbf, 128, wLK, F(LSA_KB), VK, (int)TL, 64, 128);  cvta(VK, bVK, TL*64);
  lin(Lbf, 128, wLV, F(LSA_VB), VV, (int)TL, 64, 128);
  packT_kernel<<<(unsigned)((256LL*4*16*80+255)/256),256,0,stream>>>(VV, VT, 256LL*4*16*80, 4, 16, 80, 64);
  gemm(bVQ,64, 80*64,16, bVK,64, 80*64,16, SC,80, 4LL*6400,6400, nullptr, 80,80,16, 1024,4);
  softmax_kernel<<<1024*80,256,0,stream>>>(SC, 80, 80, 0.25f, nullptr, 1, 0, lmask, 320);
  gemm((const __bf16*)SC,160, 4LL*80*160,80*160, VT,80, 4LL*1280,1280,
       VT1,64, 80*64,16, nullptr, 80,16,80, 1024,4);
  cvta(VT1, bVT1, TL*64);
  lin(bVT1, 64, wLO, F(LSA_OB), VQ, (int)TL, 128, 64);
  ln(Lf, VQ, LSA_LNG, LSA_LNB, Lf, Lbf, TL, 0);             // l2 (in place)

  // ================= phase V =================
  auto cross_attn = [&](const __bf16* vin_bf, const float* vin_f32, float* vout, __bf16* vout_bf,
                        __bf16* wq, __bf16* wk, __bf16* wv, __bf16* wo,
                        int qb, int kb, int vb, int ob, int lng, int lnb) {
    lin(vin_bf, 128, wq, F(qb), VQ, (int)TV, 128, 128);  cvta(VQ, bVQ, TV*128);
    lin(Lbf,    128, wk, F(kb), VK, (int)TL, 128, 128);  cvta(VK, bVK, TL*128);
    lin(Lbf,    128, wv, F(vb), VV, (int)TL, 128, 128);
    packT_kernel<<<(unsigned)((256LL*4*32*80+255)/256),256,0,stream>>>(VV, VT, 256LL*4*32*80, 4, 32, 80, 128);
    gemm(bVQ,128, 256*128,32, bVK,128, 80*128,32, SC,80, 4LL*20480,20480, nullptr, 256,80,32, 1024,4);
    softmax_kernel<<<1024*256,256,0,stream>>>(SC, 80, 256, SCL32, nullptr, 1, 0, lmask, 1024);
    gemm((const __bf16*)SC,160, 4LL*256*160,256*160, VT,80, 4LL*2560,2560,
         VT1,128, 256*128,32, nullptr, 256,32,80, 1024,4);
    cvta(VT1, bVT1, TV*128);
    lin(bVT1, 128, wo, F(ob), VQ, (int)TV, 128, 128);
    ln(vin_f32, VQ, lng, lnb, vout, vout_bf, TV, 0);
  };

  // ca0: v0 (VA) -> v1 (VB)
  cross_attn(bVA, VA, VB, bVB, wC0Q, wC0K, wC0V, wC0O, CA0_QB, CA0_KB, CA0_VB, CA0_OB, CA0_LNG, CA0_LNB);

  // sa1: v1 (VB) -> v2 (VA), with distance bias
  lin(bVB, 128, wS1Q, F(SA1_QB), VQ, (int)TV, 128, 128);  cvta(VQ, bVQ, TV*128);
  lin(bVB, 128, wS1K, F(SA1_KB), VK, (int)TV, 128, 128);  cvta(VK, bVK, TV*128);
  lin(bVB, 128, wS1V, F(SA1_VB), VV, (int)TV, 128, 128);
  packT_kernel<<<(unsigned)((256LL*4*32*256+255)/256),256,0,stream>>>(VV, VT, 256LL*4*32*256, 4, 32, 256, 128);
  gemm(bVQ,128, 256*128,32, bVK,128, 256*128,32, SC,256, 4LL*65536,65536, nullptr, 256,256,32, 1024,4);
  softmax_kernel<<<1024*256,256,0,stream>>>(SC, 256, 256, SCL32, DW, 32768, 65536, nullptr, 1);
  gemm((const __bf16*)SC,512, 4LL*256*512,256*512, VT,256, 4LL*8192,8192,
       VT1,128, 256*128,32, nullptr, 256,32,256, 1024,4);
  cvta(VT1, bVT1, TV*128);
  lin(bVT1, 128, wS1O, F(SA1_OB), VQ, (int)TV, 128, 128);
  ln(VB, VQ, SA1_LNG, SA1_LNB, VA, bVA, TV, 0);             // v2

  // ca1: v2 (VA) -> v3 (VB)
  cross_attn(bVA, VA, VB, bVB, wC1Q, wC1K, wC1V, wC1O, CA1_QB, CA1_KB, CA1_VB, CA1_OB, CA1_LNG, CA1_LNB);

  // ================= match head =================
  lin(bVB, 128, wM1, F(M_C1B), VT1, (int)TV, 128, 128);
  bn_prelu_kernel<<<(unsigned)((TV*128+255)/256),256,0,stream>>>(VT1, bVT1, F(M_BN1G), F(M_BN1B),
      F(M_BN1RM), F(M_BN1RV), F(M_A1), 0, TV*128, 128);
  lin(bVT1, 128, wM2, F(M_C2B), VQ, (int)TV, 128, 128);
  bn_prelu_kernel<<<(unsigned)((TV*128+255)/256),256,0,stream>>>(VQ, nullptr, F(M_BN2G), F(M_BN2B),
      F(M_BN2RM), F(M_BN2RV), F(M_A2), 0, TV*128, 128);
  dot_kernel<<<(unsigned)(TV/8),256,0,stream>>>(VQ, F(M_C3W), F(M_C3B), (float*)d_out, (int)TV);
}
